// PathRNN_39840116637913
// MI455X (gfx1250) — matrix operand order repaired and checked by reading, not compiled, not run on hardware
//
#include <hip/hip_runtime.h>
#include <hip/hip_bf16.h>

#define B_SZ 128
#define T_SZ 1024
#define H_SZ 512
#define BB 16                       // batch rows per workgroup (= WMMA M tile)
#define NWAVES 8
#define COLS_PER_WAVE (H_SZ / NWAVES)   // 64
#define NTILES (COLS_PER_WAVE / 16)     // 4 N-tiles per wave
#define KTILES (H_SZ / 32)              // 16 K-tiles (K=32 per bf16 WMMA)
#define KRES   8                        // K-tiles register-resident (256 VGPRs)
#define HPAD 8
#define LDS_STRIDE (H_SZ + HPAD)        // 520 bf16 per row (bank-conflict pad)

typedef __attribute__((ext_vector_type(16))) __bf16 v16bf;
typedef __attribute__((ext_vector_type(8)))  float  v8f;
typedef __attribute__((ext_vector_type(4)))  unsigned int u32x4;

union Frag16 { v16bf v; u32x4 u[2]; };

// ---------------------------------------------------------------------------
// Pre-swizzle W_hh (fp32, [H][H], row j = output col, col k = input) into
// bf16 WMMA B-fragment order: Wb[ct][kt][lane][i], 32 contiguous bytes/lane.
// B matrix (K=32 x N=16): lanes 0-15 hold N=lane, K=0..15 (elem i <-> K=i);
// lanes 16-31 hold N=lane-16, K=16..31.  B[k][n] = W_hh[n][k].
// ---------------------------------------------------------------------------
__global__ void PathRNN_prep_whh(const float* __restrict__ W_hh,
                                 __bf16* __restrict__ Wb) {
    int idx  = blockIdx.x * blockDim.x + threadIdx.x; // 32*16*32*16 threads
    int i    = idx & 15;
    int lane = (idx >> 4) & 31;
    int kt   = (idx >> 9) & 15;
    int ct   = idx >> 13;                    // global column tile 0..31
    int j    = ct * 16 + (lane & 15);        // output column (row of W_hh)
    int k    = kt * 32 + ((lane & 16) ? 16 : 0) + i;
    Wb[idx]  = (__bf16)W_hh[j * H_SZ + k];
}

// ---------------------------------------------------------------------------
// Main recurrence: one workgroup (8 waves) per 16-row batch block.
// Each wave owns 64 hidden columns (4 N-tiles). h kept in LDS as bf16.
// W_hh B-fragments: K-tiles 0..7 register-resident (256 VGPRs, loaded once);
// K-tiles 8..15 streamed from L2 every step.  A 32-bit opaque zero offset
// (asm-laundered per iteration) blocks LICM from hoisting the streamed loads
// while preserving global address-space inference (global_load, not flat).
// A fragments are software-pipelined one K-tile ahead; streamed B fragments
// are double-buffered one group ahead.
// ---------------------------------------------------------------------------
__global__ __launch_bounds__(256, 1)
void PathRNN_main(const float* __restrict__ vel,     // [B][T][2]
                  const float* __restrict__ W_ih,    // [H][2]
                  const float* __restrict__ b_ih,    // [H]
                  const float* __restrict__ b_hh,    // [H]
                  const __bf16* __restrict__ Wb,     // swizzled B fragments
                  float* __restrict__ u_out) {       // [B][T][H]
    __shared__ __bf16 h_lds[BB][LDS_STRIDE];

    const int tid     = threadIdx.x;
    const int wave    = tid >> 5;
    const int lane    = tid & 31;
    const int nlo     = lane & 15;            // N (col-in-tile) and A-row M
    const int hi      = (lane & 16) ? 1 : 0;  // lane half
    const int mh      = hi ? 8 : 0;           // C/D row offset for this half
    const int bbase   = blockIdx.x * BB;
    const int colbase = wave * COLS_PER_WAVE;

    // zero initial hidden state
    for (int i = tid; i < BB * LDS_STRIDE; i += 256)
        (&h_lds[0][0])[i] = (__bf16)0.0f;

    // per-lane column constants (input proj weights + biases for 4 tiles)
    float wih0[NTILES], wih1[NTILES], bias[NTILES];
#pragma unroll
    for (int ct = 0; ct < NTILES; ++ct) {
        int j    = colbase + ct * 16 + nlo;
        wih0[ct] = W_ih[j * 2 + 0];
        wih1[ct] = W_ih[j * 2 + 1];
        bias[ct] = b_ih[j] + b_hh[j];
    }

    // register-resident B fragments: K-tiles 0..KRES-1 (4ct x 8kt x 8 VGPRs)
    Frag16 bw[NTILES][KRES];
#pragma unroll
    for (int ct = 0; ct < NTILES; ++ct) {
#pragma unroll
        for (int kt = 0; kt < KRES; ++kt) {
            const char* fb = (const char*)Wb +
                (((size_t)((colbase >> 4) + ct) * KTILES + kt) * 32 + lane) * 32;
            bw[ct][kt].u[0] = *(const u32x4*)(fb);
            bw[ct][kt].u[1] = *(const u32x4*)(fb + 16);
        }
    }

    __syncthreads();

    const char* arow = (const char*)&h_lds[nlo][0];

    for (int t = 0; t < T_SZ; ++t) {
        // opaque zero offset: redefined every iteration so streamed B loads
        // cannot be hoisted, but Wb's global addrspace is preserved.
        int tzero = 0;
        asm volatile("" : "+v"(tzero));
        const char* wbs = (const char*)Wb + tzero;

        auto load_a = [&](int kt) {
            Frag16 a;   // A: M=16 x K=32 bf16; this lane = row nlo
            a.u[0] = *(const u32x4*)(arow + kt * 64 + hi * 32);
            a.u[1] = *(const u32x4*)(arow + kt * 64 + hi * 32 + 16);
            return a;
        };
        auto load_b = [&](int kt, int ct, Frag16& f) {
            const char* fb = wbs +
                (((size_t)((colbase >> 4) + ct) * KTILES + kt) * 32 + lane) * 32;
            f.u[0] = *(const u32x4*)(fb);
            f.u[1] = *(const u32x4*)(fb + 16);
        };

        // input projection -> accumulator init (C matrix)
        float2 vv[8];
#pragma unroll
        for (int r = 0; r < 8; ++r) {
            int b = bbase + mh + r;
            vv[r] = *(const float2*)&vel[((size_t)b * T_SZ + t) * 2];
        }
        v8f acc[NTILES];
#pragma unroll
        for (int ct = 0; ct < NTILES; ++ct)
#pragma unroll
            for (int r = 0; r < 8; ++r)
                acc[ct][r] = vv[r].x * wih0[ct] + vv[r].y * wih1[ct] + bias[ct];

        // prefetch first streamed B group (kt = KRES) and first A fragment
        Frag16 sb[2][NTILES];
#pragma unroll
        for (int ct = 0; ct < NTILES; ++ct) load_b(KRES, ct, sb[0][ct]);
        Frag16 a_cur = load_a(0);

        // unified K loop: A pipelined 1 tile ahead, streamed B 1 group ahead
#pragma unroll
        for (int kt = 0; kt < KTILES; ++kt) {
            Frag16 a_nxt;
            if (kt + 1 < KTILES) a_nxt = load_a(kt + 1);
            if (kt >= KRES && kt + 1 < KTILES) {
                const int gn = kt - KRES + 1;
#pragma unroll
                for (int ct = 0; ct < NTILES; ++ct)
                    load_b(kt + 1, ct, sb[gn & 1][ct]);
            }
            if (kt < KRES) {
#pragma unroll
                for (int ct = 0; ct < NTILES; ++ct)
                    acc[ct] = __builtin_amdgcn_wmma_f32_16x16x32_bf16(
                        false, a_cur.v, false, bw[ct][kt].v, (short)0, acc[ct],
                        false, false);
            } else {
                const int g = kt - KRES;
#pragma unroll
                for (int ct = 0; ct < NTILES; ++ct)
                    acc[ct] = __builtin_amdgcn_wmma_f32_16x16x32_bf16(
                        false, a_cur.v, false, sb[g & 1][ct].v, (short)0, acc[ct],
                        false, false);
            }
            if (kt + 1 < KTILES) a_cur = a_nxt;
        }

        __syncthreads();   // everyone done reading h_{t-1}

        // tanh, write u (fp32, global) and h_t (bf16, LDS)
#pragma unroll
        for (int ct = 0; ct < NTILES; ++ct) {
            int j = colbase + ct * 16 + nlo;
#pragma unroll
            for (int r = 0; r < 8; ++r) {
                float x = acc[ct][r];
                float e = __expf(2.0f * x);
                float u = 1.0f - 2.0f / (e + 1.0f);     // tanh(x), saturating
                int b = bbase + mh + r;
                u_out[((size_t)b * T_SZ + t) * H_SZ + j] = u;
                h_lds[mh + r][j] = (__bf16)u;
            }
        }
        __syncthreads();   // h_t visible before next step's reads
    }
}

// ---------------------------------------------------------------------------
// Output projection: pos[b,t,o] = sum_h u[b,t,h] * W_out[o,h] + b_out[o].
// One wave per (b,t); lane covers 16 of 512 h; shuffle reduction.
// ---------------------------------------------------------------------------
__global__ __launch_bounds__(256)
void PathRNN_proj(const float* __restrict__ u,       // [B*T][H]
                  const float* __restrict__ W_out,   // [2][H]
                  const float* __restrict__ b_out,   // [2]
                  float* __restrict__ pos) {         // [B*T][2]
    int gw   = (blockIdx.x * blockDim.x + threadIdx.x) >> 5; // (b*T + t)
    int lane = threadIdx.x & 31;
    const float* urow = u + (size_t)gw * H_SZ + lane * 16;
    float s0 = 0.f, s1 = 0.f;
#pragma unroll
    for (int i = 0; i < 16; i += 4) {
        float4 uv = *(const float4*)(urow + i);
        int j = lane * 16 + i;
        float4 w0 = *(const float4*)(W_out + j);
        float4 w1 = *(const float4*)(W_out + H_SZ + j);
        s0 += uv.x * w0.x + uv.y * w0.y + uv.z * w0.z + uv.w * w0.w;
        s1 += uv.x * w1.x + uv.y * w1.y + uv.z * w1.z + uv.w * w1.w;
    }
#pragma unroll
    for (int off = 16; off >= 1; off >>= 1) {
        s0 += __shfl_xor(s0, off, 32);
        s1 += __shfl_xor(s1, off, 32);
    }
    if (lane == 0) {
        pos[(size_t)gw * 2 + 0] = s0 + b_out[0];
        pos[(size_t)gw * 2 + 1] = s1 + b_out[1];
    }
}

extern "C" void kernel_launch(void* const* d_in, const int* in_sizes, int n_in,
                              void* d_out, int out_size, void* d_ws, size_t ws_size,
                              hipStream_t stream) {
    const float* vel   = (const float*)d_in[0];
    const float* W_ih  = (const float*)d_in[1];
    const float* W_hh  = (const float*)d_in[2];
    const float* b_ih  = (const float*)d_in[3];
    const float* b_hh  = (const float*)d_in[4];
    const float* W_out = (const float*)d_in[5];
    const float* b_out = (const float*)d_in[6];

    float* pos = (float*)d_out;                         // [B,T,2] first
    float* u   = pos + (size_t)B_SZ * T_SZ * 2;         // then [B,T,H]
    __bf16* Wb = (__bf16*)d_ws;                         // 512 KB swizzled W_hh

    // 1) swizzle W_hh -> bf16 B-fragments (32ct*16kt*32lane*16 elems)
    PathRNN_prep_whh<<<(32 * 16 * 32 * 16) / 256, 256, 0, stream>>>(W_hh, Wb);

    // 2) recurrence: 8 batch blocks x 8 waves
    PathRNN_main<<<B_SZ / BB, NWAVES * 32, 0, stream>>>(vel, W_ih, b_ih, b_hh, Wb, u);

    // 3) output projection: one wave per (b,t)
    PathRNN_proj<<<(B_SZ * T_SZ) / 8, 256, 0, stream>>>(u, W_out, b_out, pos);
}
